// LocalFeatue_82154134438331
// MI455X (gfx1250) — compile-verified, matrix operand order checked by reading, not run on hardware
//
#include <hip/hip_runtime.h>
#include <math.h>

// ---- problem constants ----
#define BATCH 2
#define NPTS  4096
#define KNBR  32
#define MTOT  (BATCH * NPTS * KNBR)   // 262144 points through the GEMMs
#define NTILES (MTOT / 16)            // 16384 16-point tiles
#define TM    2                       // point-tiles per wave
#define NWAVES (NTILES / TM)          // 8192 gemm waves
#define WPB   (NWAVES / BATCH)        // 4096 waves per batch

// ---- workspace layout (bytes) ----
#define X0_OFF 0u                                  // 16384*512 halfs  = 16 MB (13ch padded to 32)
#define Y0_OFF (16777216u)                         // 262144*64 halfs  = 33.5 MB
#define Y1_OFF (Y0_OFF + 33554432u)
#define Y2_OFF (Y1_OFF + 33554432u)                // 262144*128 halfs = 67 MB
#define W_OFF  (Y2_OFF + 67108864u)                // 14336 halfs (W0 pad 64x32, W1 64x64, W2 128x64)
#define ST_OFF (W_OFF + 32768u)                    // 3 layers * 2 b * 8 g * 2 (sum,sumsq) f32
#define AF_OFF (ST_OFF + 512u)                     // 3 layers * 2 b * 128 c * 2 (a,b) f32
#define PART_OFF (AF_OFF + 8192u)                  // 8192 waves * 8 otiles * 32 lanes * float2 = 16 MB

typedef __attribute__((ext_vector_type(16))) _Float16 v16h;
typedef __attribute__((ext_vector_type(8)))  _Float16 v8h;
typedef __attribute__((ext_vector_type(8)))  float    v8f;

// -------------------- small helpers --------------------
__device__ __forceinline__ float angf(float ax, float ay, float az,
                                      float bx, float by, float bz) {
  float cx = ay * bz - az * by;
  float cy = az * bx - ax * bz;
  float cz = ax * by - ay * bx;
  float s = cx * cx + cy * cy + cz * cz;
  float y = s > 0.f ? sqrtf(s) : 0.f;
  float x = ax * bx + ay * by + az * bz;
  if (y == 0.f && x == 0.f) return 0.f;
  return atan2f(y, x);
}

// -------------------- weight prep: f32 -> f16, pad layer0 to 32 cols ----
__global__ void prep_w_kernel(const float* __restrict__ w0,
                              const float* __restrict__ w1,
                              const float* __restrict__ w2,
                              _Float16* __restrict__ Wf) {
  int i = blockIdx.x * 256 + threadIdx.x;
  if (i < 2048) {                       // W0: 64 rows x 32 cols (src 64x13)
    int r = i >> 5, c = i & 31;
    Wf[i] = (_Float16)(c < 13 ? w0[r * 13 + c] : 0.f);
  } else if (i < 6144) {                // W1: 64x64
    Wf[i] = (_Float16)w1[i - 2048];
  } else if (i < 14336) {               // W2: 128x64
    Wf[i] = (_Float16)w2[i - 6144];
  }
}

// -------------------- ball grouping + 13-ch fused features --------------
__global__ void group_feat_kernel(const float* __restrict__ feat,
                                  const float* __restrict__ xyz,
                                  const float* __restrict__ wa,
                                  const float* __restrict__ wb,
                                  const float* __restrict__ wc,
                                  _Float16* __restrict__ X0) {
  __shared__ int sidx[8][32];
  const int lane = threadIdx.x & 31;
  const int wid  = threadIdx.x >> 5;
  const int p = blockIdx.x * 8 + wid;        // 0 .. B*N-1
  const int b = p >> 12;
  const int n = p & 4095;
  const float* xb = xyz  + (size_t)b * NPTS * 3;
  const float* fb = feat + (size_t)b * NPTS * 3;

  const float cx = xb[n * 3 + 0], cy = xb[n * 3 + 1], cz = xb[n * 3 + 2];
  const float r2 = 0.1f * 0.1f;

  int found = 0;
  for (int c0 = 0; c0 < NPTS && found < 32; c0 += 32) {
    int m = c0 + lane;
    float dx = xb[m * 3 + 0] - cx;
    float dy = xb[m * 3 + 1] - cy;
    float dz = xb[m * 3 + 2] - cz;
    bool hit = (dx * dx + dy * dy + dz * dz) <= r2;
    unsigned mask = (unsigned)__ballot(hit);
    int prefix = __popc(mask & ((1u << lane) - 1u));
    if (hit) {
      int slot = found + prefix;
      if (slot < 32) sidx[wid][slot] = m;
    }
    found += __popc(mask);
  }
  if (found > 32) found = 32;
  int first = sidx[wid][0];                 // self is always in-radius
  if (lane >= found) sidx[wid][lane] = first;

  const int kn = lane;
  const int id = sidx[wid][kn];
  float gxa = xb[id * 3 + 0], gya = xb[id * 3 + 1], gza = xb[id * 3 + 2];
  float g1x = gxa - cx, g1y = gya - cy, g1z = gza - cz;
  float sq = g1x * g1x + g1y * g1y + g1z * g1z;
  float dist = sq > 0.f ? sqrtf(sq) : 0.f;
  float nrx = fb[n * 3 + 0],  nry = fb[n * 3 + 1],  nrz = fb[n * 3 + 2];
  float gnx = fb[id * 3 + 0], gny = fb[id * 3 + 1], gnz = fb[id * 3 + 2];
  float a0 = angf(nrx, nry, nrz, g1x, g1y, g1z);
  float a1 = angf(gnx, gny, gnz, g1x, g1y, g1z);
  float a2 = angf(nrx, nry, nrz, gnx, gny, gnz);

  const float wav = wa[0], wbv = wb[0], wcv = wc[0];
  float v[13];
  v[0] = wav * cx;  v[1] = wav * cy;  v[2] = wav * cz;
  v[3] = wav * gxa; v[4] = wav * gya; v[5] = wav * gza;
  v[6] = wbv * g1x; v[7] = wbv * g1y; v[8] = wbv * g1z;
  v[9] = wbv * dist;
  v[10] = wcv * a0; v[11] = wcv * a1; v[12] = wcv * a2;

  size_t t = (size_t)p * 2 + (kn >> 4);
  int j = kn & 15;
  _Float16* dst = X0 + t * 512 + (size_t)j * 32;
  #pragma unroll
  for (int c = 0; c < 13; ++c) dst[c] = (_Float16)v[c];
  #pragma unroll
  for (int c = 13; c < 32; ++c) dst[c] = (_Float16)0.f;
}

// -------------------- WMMA GEMM + coalesced stats-partial epilogue ------
// One wave handles TM=2 consecutive 16-point tiles: weight A-fragments are
// loaded once per o-tile and reused for both B-tiles. GN stats leave the
// kernel as per-lane float2 partials (one coalesced 8B store per o-tile) —
// no shuffles, no atomics on the WMMA critical path.
template <int NKSTEP, int GSIZE, bool USEAFF, int OC>
__global__ void gemm_kernel(const _Float16* __restrict__ Xin,
                            const _Float16* __restrict__ Wt,
                            const float* __restrict__ affIn,
                            _Float16* __restrict__ Yout,
                            float2* __restrict__ part) {
  constexpr int CIN = NKSTEP * 32;              // stored channel stride
  constexpr int NOTILE = OC / 16;
  const int lane = threadIdx.x & 31;
  const int wid  = threadIdx.x >> 5;
  const int w    = blockIdx.x * 8 + wid;        // gemm-wave id < 8192
  const int t0   = w * TM;                      // first tile
  const int h    = lane >> 4;                   // half select
  const int nl   = lane & 15;                   // point within tile / A row
  const int bat  = w >> 12;                     // batch (4096 waves each)

  // ---- B fragments: TM tiles x NKSTEP k-steps, two 16B loads each -------
  v16h bf[TM][NKSTEP];
  #pragma unroll
  for (int m = 0; m < TM; ++m) {
    #pragma unroll
    for (int s = 0; s < NKSTEP; ++s) {
      const _Float16* src = Xin + ((size_t)(t0 + m) * 16 + nl) * CIN + s * 32;
      v8h lo = *(const v8h*)(src + 8 * h);      // K = s*32 + 8h   .. +7
      v8h hi = *(const v8h*)(src + 16 + 8 * h); // K = s*32+16+8h .. +7
      if (USEAFF) {
        const float* af = affIn + (size_t)bat * 256;
        float alo[16], ahi[16];                 // (a,b) pairs, 64B each chunk
        const float4* plo = (const float4*)(af + (s * 32 + 8 * h) * 2);
        const float4* phi = (const float4*)(af + (s * 32 + 16 + 8 * h) * 2);
        #pragma unroll
        for (int q = 0; q < 4; ++q) {
          *(float4*)&alo[q * 4] = plo[q];
          *(float4*)&ahi[q * 4] = phi[q];
        }
        #pragma unroll
        for (int e = 0; e < 8; ++e) {
          float x = (float)lo[e] * alo[2 * e] + alo[2 * e + 1];
          lo[e] = (_Float16)(x > 0.f ? x : 0.f);
          float y = (float)hi[e] * ahi[2 * e] + ahi[2 * e + 1];
          hi[e] = (_Float16)(y > 0.f ? y : 0.f);
        }
      }
      *(v8h*)&bf[m][s] = lo;
      *(((v8h*)&bf[m][s]) + 1) = hi;
    }
  }

  #pragma unroll
  for (int o = 0; o < NOTILE; ++o) {
    // A fragments for this o-tile (shared by both B-tiles)
    v16h afr[NKSTEP];
    #pragma unroll
    for (int s = 0; s < NKSTEP; ++s) {
      const _Float16* wr = Wt + (size_t)(o * 16 + nl) * CIN + s * 32;
      *(v8h*)&afr[s] = *(const v8h*)(wr + 8 * h);
      *(((v8h*)&afr[s]) + 1) = *(const v8h*)(wr + 16 + 8 * h);
    }
    float s1 = 0.f, s2 = 0.f;
    #pragma unroll
    for (int m = 0; m < TM; ++m) {
      v8f acc = {};
      #pragma unroll
      for (int s = 0; s < NKSTEP; ++s)
        acc = __builtin_amdgcn_wmma_f32_16x16x32_f16(
            false, afr[s], false, bf[m][s], (short)0, acc, false, false);
      // store D: lane holds out-channels c = o*16 + 8h + j  (j = 0..7)
      _Float16* dst =
          Yout + ((size_t)(t0 + m) * 16 + nl) * OC + o * 16 + 8 * h;
      v8h oh;
      #pragma unroll
      for (int j = 0; j < 8; ++j) {
        float x = acc[j];
        s1 += x;
        s2 += x * x;
        oh[j] = (_Float16)x;
      }
      *(v8h*)dst = oh;
    }
    // per-lane GN partial: all 8 channels of this lane are in one group
    part[((size_t)w * NOTILE + o) * 32 + lane] = make_float2(s1, s2);
  }
  (void)GSIZE;
}

// -------------------- deterministic stats reduction ----------------------
// One block per (batch, group): tree-sum the per-lane float2 partials.
__global__ void stats_reduce_kernel(const float2* __restrict__ part,
                                    float* __restrict__ statsL, int notile,
                                    int gsize) {
  const int bat = blockIdx.x >> 3;
  const int g   = blockIdx.x & 7;
  float s1 = 0.f, s2 = 0.f;
  if (gsize == 8) {                 // group = 2o + h : o = g>>1, lanes h*16..
    const int o = g >> 1;
    const int laneBase = (g & 1) * 16;
    for (int idx = threadIdx.x; idx < WPB * 16; idx += 256) {
      int w = bat * WPB + (idx >> 4);
      int lane = laneBase + (idx & 15);
      float2 v = part[((size_t)w * notile + o) * 32 + lane];
      s1 += v.x;
      s2 += v.y;
    }
  } else {                          // 16-channel groups: group = o, all lanes
    const int o = g;
    for (int idx = threadIdx.x; idx < WPB * 32; idx += 256) {
      int w = bat * WPB + (idx >> 5);
      int lane = idx & 31;
      float2 v = part[((size_t)w * notile + o) * 32 + lane];
      s1 += v.x;
      s2 += v.y;
    }
  }
  __shared__ float rs[256], rq[256];
  rs[threadIdx.x] = s1;
  rq[threadIdx.x] = s2;
  __syncthreads();
  for (int st = 128; st > 0; st >>= 1) {
    if (threadIdx.x < st) {
      rs[threadIdx.x] += rs[threadIdx.x + st];
      rq[threadIdx.x] += rq[threadIdx.x + st];
    }
    __syncthreads();
  }
  if (threadIdx.x == 0) {
    statsL[((size_t)bat * 8 + g) * 2 + 0] = rs[0];
    statsL[((size_t)bat * 8 + g) * 2 + 1] = rq[0];
  }
}

// -------------------- stats -> per-channel affine (a, b) -----------------
__global__ void affine_kernel(const float* __restrict__ stats,
                              const float* __restrict__ gamma,
                              const float* __restrict__ beta,
                              float* __restrict__ aff, int C, float invCount) {
  int i = blockIdx.x * blockDim.x + threadIdx.x;
  if (i >= 2 * C) return;
  int bat = i / C, c = i % C;
  int cg = (C == 128) ? 16 : 8;
  int g = c / cg;
  float sum = stats[((size_t)bat * 8 + g) * 2 + 0];
  float sq  = stats[((size_t)bat * 8 + g) * 2 + 1];
  float mean = sum * invCount;
  float var  = sq * invCount - mean * mean;
  float r = rsqrtf(var + 1e-5f);
  float a = gamma[c] * r;
  float bb = beta[c] - mean * a;
  aff[((size_t)bat * 128 + c) * 2 + 0] = a;
  aff[((size_t)bat * 128 + c) * 2 + 1] = bb;
}

// -------------------- final: GN-affine + max over K (relu'd and raw) -----
__global__ void final_kernel(const _Float16* __restrict__ Y2,
                             const float* __restrict__ aff2,
                             float* __restrict__ out) {
  int i = blockIdx.x * 256 + threadIdx.x;     // 2*128*4096 threads
  int c = i & 127;
  int n = (i >> 7) & 4095;
  int b = i >> 19;
  float a  = aff2[((size_t)b * 128 + c) * 2 + 0];
  float bb = aff2[((size_t)b * 128 + c) * 2 + 1];
  size_t base = ((size_t)(b * NPTS + n)) * 32;   // 32 consecutive pt rows
  const _Float16* p0 = Y2 + base * 128 + c;
  float mx = -3.4e38f;
  #pragma unroll 8
  for (int j = 0; j < 32; ++j) {
    float v = (float)p0[(size_t)j * 128];
    v = v * a + bb;
    mx = fmaxf(mx, v);
  }
  out[((size_t)b * 128 + c) * 4096 + n] = fmaxf(mx, 0.f);   // max(relu(x))
  out[1048576u + ((size_t)b * 128 + c) * 4096 + n] = mx;    // max(x)
}

// -------------------- launcher --------------------
extern "C" void kernel_launch(void* const* d_in, const int* in_sizes, int n_in,
                              void* d_out, int out_size, void* d_ws,
                              size_t ws_size, hipStream_t stream) {
  (void)in_sizes; (void)n_in; (void)out_size; (void)ws_size;
  const float* feat = (const float*)d_in[0];
  const float* xyz  = (const float*)d_in[1];
  const float* w0 = (const float*)d_in[2];
  const float* g0 = (const float*)d_in[3];
  const float* b0 = (const float*)d_in[4];
  const float* w1 = (const float*)d_in[5];
  const float* g1 = (const float*)d_in[6];
  const float* b1 = (const float*)d_in[7];
  const float* w2 = (const float*)d_in[8];
  const float* g2 = (const float*)d_in[9];
  const float* b2 = (const float*)d_in[10];
  const float* wa = (const float*)d_in[11];
  const float* wb = (const float*)d_in[12];
  const float* wc = (const float*)d_in[13];

  char* ws = (char*)d_ws;
  _Float16* X0 = (_Float16*)(ws + X0_OFF);
  _Float16* Y0 = (_Float16*)(ws + Y0_OFF);
  _Float16* Y1 = (_Float16*)(ws + Y1_OFF);
  _Float16* Y2 = (_Float16*)(ws + Y2_OFF);
  _Float16* Wf = (_Float16*)(ws + W_OFF);
  float*    ST = (float*)(ws + ST_OFF);   // [3][2][8][2]
  float*    AF = (float*)(ws + AF_OFF);   // [3][2][128][2]
  float2*   PT = (float2*)(ws + PART_OFF);
  float*    out = (float*)d_out;

  _Float16* W0f = Wf;          // 64 x 32
  _Float16* W1f = Wf + 2048;   // 64 x 64
  _Float16* W2f = Wf + 6144;   // 128 x 64
  float* ST0 = ST;       float* ST1 = ST + 32;  float* ST2 = ST + 64;
  float* AF0 = AF;       float* AF1 = AF + 512; float* AF2 = AF + 1024;

  prep_w_kernel<<<(14336 + 255) / 256, 256, 0, stream>>>(w0, w1, w2, Wf);
  group_feat_kernel<<<(BATCH * NPTS) / 8, 256, 0, stream>>>(feat, xyz, wa, wb,
                                                            wc, X0);
  // layer 0: C=13(pad32), O=64, groups of 8
  gemm_kernel<1, 8, false, 64><<<NWAVES / 8, 256, 0, stream>>>(X0, W0f,
                                                               nullptr, Y0, PT);
  stats_reduce_kernel<<<16, 256, 0, stream>>>(PT, ST0, 4, 8);
  affine_kernel<<<1, 128, 0, stream>>>(ST0, g0, b0, AF0, 64,
                                       1.f / (8.f * 32.f * 4096.f));
  // layer 1: C=64, O=64, groups of 8
  gemm_kernel<2, 8, true, 64><<<NWAVES / 8, 256, 0, stream>>>(Y0, W1f, AF0, Y1,
                                                              PT);
  stats_reduce_kernel<<<16, 256, 0, stream>>>(PT, ST1, 4, 8);
  affine_kernel<<<1, 128, 0, stream>>>(ST1, g1, b1, AF1, 64,
                                       1.f / (8.f * 32.f * 4096.f));
  // layer 2: C=64, O=128, groups of 16
  gemm_kernel<2, 16, true, 128><<<NWAVES / 8, 256, 0, stream>>>(Y1, W2f, AF1,
                                                                Y2, PT);
  stats_reduce_kernel<<<16, 256, 0, stream>>>(PT, ST2, 8, 16);
  affine_kernel<<<1, 256, 0, stream>>>(ST2, g2, b2, AF2, 128,
                                       1.f / (16.f * 32.f * 4096.f));
  final_kernel<<<1048576 / 256, 256, 0, stream>>>(Y2, AF2, out);
}
